// HierarchicalPathNetworkLayer_31980326486649
// MI455X (gfx1250) — compile-verified, matrix operand order checked by reading, not run on hardware
//
#include <hip/hip_runtime.h>
#include <hip/hip_bf16.h>

// ---------------------------------------------------------------------------
// CDNA5 (gfx1250) implementation of HierarchicalPathNetworkLayer.
//   - MLPs + final projection: V_WMMA_F32_16X16X32_BF16, one 16-row tile per
//     wave32; weights staged into LDS via GLOBAL_LOAD_ASYNC_TO_LDS_B128
//     (ASYNCcnt) with bank-conflict-free padded rows; hidden layer staged in
//     LDS (never touches HBM).
//   - segment_prod / segment_sum: bandwidth-bound float4 kernels, scatter via
//     global_atomic_add_f32.
// ---------------------------------------------------------------------------

typedef __attribute__((ext_vector_type(16))) __bf16 v16bf;
typedef __attribute__((ext_vector_type(8)))  float  v8f;
typedef __attribute__((ext_vector_type(4)))  int    v4i;

#define WMMA_BF16(a, b, c) \
  __builtin_amdgcn_wmma_f32_16x16x32_bf16(false, (a), false, (b), (short)0, (c), false, false)

// Padded LDS row stride (bf16 elements): 136*2B = 272B = 17 x 16B
//  -> ds_load_b128-aligned, +4 banks per row => conflict-free half-wave reads.
#define LDS_STRIDE 136

#if defined(__gfx1250__) && \
    __has_builtin(__builtin_amdgcn_global_load_async_to_lds_b128) && \
    __has_builtin(__builtin_amdgcn_s_wait_asynccnt)
#define HAVE_ASYNC_LDS 1
#else
#define HAVE_ASYNC_LDS 0
#endif

static __device__ __forceinline__ float silu(float x) {
  return x / (1.0f + __expf(-x));
}

// Stage a 128x128 bf16 weight tile (row stride gstride in elements) from
// global into LDS with padded rows. One b128 (8 bf16) chunk per iteration.
static __device__ __forceinline__ void stage_w128(const __bf16* __restrict__ g,
                                                  int gstride,
                                                  __bf16* __restrict__ l) {
  for (int i = threadIdx.x; i < 128 * 16; i += blockDim.x) {
    int r = i >> 4;        // weight row (output column n)
    int c = i & 15;        // 8-element chunk within row
    const __bf16* s = g + (size_t)r * gstride + c * 8;
    __bf16* d = l + r * LDS_STRIDE + c * 8;
#if HAVE_ASYNC_LDS
    __builtin_amdgcn_global_load_async_to_lds_b128(
        (__attribute__((address_space(1))) v4i*)(size_t)s,
        (__attribute__((address_space(3))) v4i*)(unsigned)(size_t)d, 0, 0);
#else
    *(uint4*)d = *(const uint4*)s;
#endif
  }
#if HAVE_ASYNC_LDS
  __builtin_amdgcn_s_wait_asynccnt(0);
#endif
}

// A-fragment (bf16 16x32) from an fp32 row-major source row.
// lane: M = lane&15, khalf = lane>>4.
//   chunk0 = K [kb + 8*khalf, +8),  chunk1 = K [kb + 16 + 8*khalf, +8)
static __device__ __forceinline__ v16bf frag_from_f32(const float* __restrict__ row,
                                                      int kb, int khalf) {
  const float* p0 = row + kb + 8 * khalf;
  const float* p1 = row + kb + 16 + 8 * khalf;
  v16bf a;
#pragma unroll
  for (int i = 0; i < 8; ++i) a[i] = (__bf16)p0[i];
#pragma unroll
  for (int i = 0; i < 8; ++i) a[8 + i] = (__bf16)p1[i];
  return a;
}

// A/B-fragment from a bf16 row-major source row (global or LDS).
static __device__ __forceinline__ v16bf frag_from_bf16(const __bf16* row,
                                                       int kb, int khalf) {
  const __bf16* p0 = row + kb + 8 * khalf;
  const __bf16* p1 = row + kb + 16 + 8 * khalf;
  v16bf a;
#pragma unroll
  for (int i = 0; i < 8; ++i) a[i] = p0[i];
#pragma unroll
  for (int i = 0; i < 8; ++i) a[8 + i] = p1[i];
  return a;
}

// --------------------------------------------------------------------------
// Weight prep: Wt[n][k] = bf16(W[k][n])   (W is K x N row-major fp32)
// --------------------------------------------------------------------------
__global__ void wT_bf16_kernel(const float* __restrict__ W, __bf16* __restrict__ Wt,
                               int K, int N) {
  int idx = blockIdx.x * blockDim.x + threadIdx.x;
  if (idx >= K * N) return;
  int n = idx / K;
  int k = idx - n * K;
  Wt[idx] = (__bf16)W[(size_t)k * N + n];
}

// --------------------------------------------------------------------------
// Fused MLP: Y = silu(X @ Wa + ba) @ Wb + bb   (D=128)
// WaT/WbT are bf16, transposed: [N=128][K=128] row-major.
// One wave per 16-row tile; 4 waves per block. Weights staged into one shared
// LDS buffer in two phases (Wa for GEMM1, Wb for GEMM2).
// --------------------------------------------------------------------------
__global__ void __launch_bounds__(128) mlp_kernel(
    const float* __restrict__ X,
    const __bf16* __restrict__ WaT, const float* __restrict__ ba,
    const __bf16* __restrict__ WbT, const float* __restrict__ bb,
    float* __restrict__ Y, int nrows) {
  __shared__ __bf16 lds_w[128 * LDS_STRIDE];       // staged weight matrix
  __shared__ __bf16 lds_h[4 * 16 * LDS_STRIDE];    // per-wave hidden tile

  const int lane   = threadIdx.x & 31;
  const int wid    = threadIdx.x >> 5;
  const int tile   = blockIdx.x * 4 + wid;
  const bool active = (tile * 16 < nrows);         // wave-uniform
  const int m      = lane & 15;
  const int khalf  = lane >> 4;

  // ---- phase 1: stage Wa, GEMM1 ----
  stage_w128(WaT, 128, lds_w);
  __syncthreads();

  v8f zero = {};
  v8f acc[8];
#pragma unroll
  for (int t = 0; t < 8; ++t) acc[t] = zero;

  if (active) {
    const float* xrow = X + (size_t)(tile * 16 + m) * 128;
#pragma unroll
    for (int kb = 0; kb < 128; kb += 32) {
      v16bf a = frag_from_f32(xrow, kb, khalf);
#pragma unroll
      for (int t = 0; t < 8; ++t) {
        v16bf b = frag_from_bf16(lds_w + (t * 16 + m) * LDS_STRIDE, kb, khalf);
        acc[t] = WMMA_BF16(a, b, acc[t]);
      }
    }
    // bias + SiLU, stash hidden tile in LDS (C-layout -> row-major bf16)
    __bf16* h = lds_h + wid * 16 * LDS_STRIDE;
#pragma unroll
    for (int t = 0; t < 8; ++t) {
      float bv = ba[t * 16 + m];
#pragma unroll
      for (int r = 0; r < 8; ++r) {
        float v = silu(acc[t][r] + bv);
        h[(r + 8 * khalf) * LDS_STRIDE + t * 16 + m] = (__bf16)v;
      }
    }
  }

  // ---- phase 2: stage Wb, GEMM2 ----
  __syncthreads();                                 // everyone done with Wa
  stage_w128(WbT, 128, lds_w);
  __syncthreads();

  if (active) {
    v8f acc2[8];
#pragma unroll
    for (int t = 0; t < 8; ++t) acc2[t] = zero;

    const __bf16* hrow = lds_h + wid * 16 * LDS_STRIDE + m * LDS_STRIDE;
#pragma unroll
    for (int kb = 0; kb < 128; kb += 32) {
      v16bf a = frag_from_bf16(hrow, kb, khalf);
#pragma unroll
      for (int t = 0; t < 8; ++t) {
        v16bf b = frag_from_bf16(lds_w + (t * 16 + m) * LDS_STRIDE, kb, khalf);
        acc2[t] = WMMA_BF16(a, b, acc2[t]);
      }
    }

#pragma unroll
    for (int t = 0; t < 8; ++t) {
      float bv = bb[t * 16 + m];
#pragma unroll
      for (int r = 0; r < 8; ++r) {
        Y[(size_t)(tile * 16 + r + 8 * khalf) * 128 + t * 16 + m] = acc2[t][r] + bv;
      }
    }
  }
}

// --------------------------------------------------------------------------
// Final projection: out = silu(concat(h1, h2n1, hdn) @ Wo + bo)
// WoT is bf16 [OUT=128][K=384] row-major, staged in 3 phases of 128 K-cols.
// --------------------------------------------------------------------------
__global__ void __launch_bounds__(128) outproj_kernel(
    const float* __restrict__ h1, const float* __restrict__ h2n1,
    const float* __restrict__ hdn,
    const __bf16* __restrict__ WoT, const float* __restrict__ bo,
    float* __restrict__ out, int nrows) {
  __shared__ __bf16 lds_w[128 * LDS_STRIDE];

  const int lane   = threadIdx.x & 31;
  const int wid    = threadIdx.x >> 5;
  const int tile   = blockIdx.x * 4 + wid;
  const bool active = (tile * 16 < nrows);
  const int m      = lane & 15;
  const int khalf  = lane >> 4;
  const size_t rowoff = (size_t)(tile * 16 + m) * 128;

  v8f zero = {};
  v8f acc[8];
#pragma unroll
  for (int t = 0; t < 8; ++t) acc[t] = zero;

  const float* srcs[3] = {h1 + rowoff, h2n1 + rowoff, hdn + rowoff};
#pragma unroll
  for (int s = 0; s < 3; ++s) {
    __syncthreads();                          // protect previous phase reads
    stage_w128(WoT + s * 128, 384, lds_w);    // K-columns [s*128, s*128+128)
    __syncthreads();
    if (active) {
      const float* xrow = srcs[s];
#pragma unroll
      for (int kb = 0; kb < 128; kb += 32) {
        v16bf a = frag_from_f32(xrow, kb, khalf);
#pragma unroll
        for (int t = 0; t < 8; ++t) {
          v16bf b = frag_from_bf16(lds_w + (t * 16 + m) * LDS_STRIDE, kb, khalf);
          acc[t] = WMMA_BF16(a, b, acc[t]);
        }
      }
    }
  }

  if (active) {
#pragma unroll
    for (int t = 0; t < 8; ++t) {
      float bv = bo[t * 16 + m];
#pragma unroll
      for (int r = 0; r < 8; ++r) {
        out[(size_t)(tile * 16 + r + 8 * khalf) * 128 + t * 16 + m] =
            silu(acc[t][r] + bv);
      }
    }
  }
}

// --------------------------------------------------------------------------
// segment_prod with exactly-2 sorted children: hout[i] = hin[src[2i]]*hin[src[2i+1]]
// --------------------------------------------------------------------------
__global__ void seg_prod2_kernel(const float* __restrict__ hin,
                                 const int* __restrict__ src,
                                 float* __restrict__ hout, int nseg) {
  int idx = blockIdx.x * blockDim.x + threadIdx.x;
  if (idx >= nseg * 32) return;
  int i = idx >> 5;
  int c = idx & 31;
  int s0 = src[2 * i];
  int s1 = src[2 * i + 1];
  float4 va = ((const float4*)(hin + (size_t)s0 * 128))[c];
  float4 vb = ((const float4*)(hin + (size_t)s1 * 128))[c];
  float4 r = make_float4(va.x * vb.x, va.y * vb.y, va.z * vb.z, va.w * vb.w);
  ((float4*)(hout + (size_t)i * 128))[c] = r;
}

// --------------------------------------------------------------------------
// scatter-add (segment_sum over reversed edges): out[src[e]] += hin[e/2]
// --------------------------------------------------------------------------
__global__ void scatter_add_kernel(const float* __restrict__ hin,
                                   const int* __restrict__ src,
                                   float* __restrict__ out, int nedges) {
  int idx = blockIdx.x * blockDim.x + threadIdx.x;
  if (idx >= nedges * 32) return;
  int e = idx >> 5;
  int c = idx & 31;
  int d = src[e];
  float4 v = ((const float4*)(hin + (size_t)(e >> 1) * 128))[c];
  float* o = out + (size_t)d * 128 + c * 4;
  atomicAdd(o + 0, v.x);
  atomicAdd(o + 1, v.y);
  atomicAdd(o + 2, v.z);
  atomicAdd(o + 3, v.w);
}

__global__ void zero_kernel(float* __restrict__ p, size_t n) {
  size_t idx = (size_t)blockIdx.x * blockDim.x + threadIdx.x;
  if (idx < n) p[idx] = 0.0f;
}

// --------------------------------------------------------------------------
// Host orchestration
// --------------------------------------------------------------------------
extern "C" void kernel_launch(void* const* d_in, const int* in_sizes, int n_in,
                              void* d_out, int out_size, void* d_ws, size_t ws_size,
                              hipStream_t stream) {
  (void)n_in; (void)ws_size; (void)out_size;
  const float* feat = (const float*)d_in[0];
  const float* W1a = (const float*)d_in[1];  const float* b1a = (const float*)d_in[2];
  const float* W1b = (const float*)d_in[3];  const float* b1b = (const float*)d_in[4];
  const float* W2a = (const float*)d_in[5];  const float* b2a = (const float*)d_in[6];
  const float* W2b = (const float*)d_in[7];  const float* b2b = (const float*)d_in[8];
  const float* W3a = (const float*)d_in[9];  const float* b3a = (const float*)d_in[10];
  const float* W3b = (const float*)d_in[11]; const float* b3b = (const float*)d_in[12];
  const float* Wo  = (const float*)d_in[13]; const float* bo  = (const float*)d_in[14];
  const int* src12 = (const int*)d_in[15];
  const int* src23 = (const int*)d_in[17];
  const int* src34 = (const int*)d_in[19];

  const int N1 = in_sizes[0] / 128;
  const int N2 = in_sizes[16] / 2;
  const int N3 = in_sizes[18] / 2;
  const int N4 = in_sizes[20] / 2;

  // ---- carve workspace ----
  size_t off = 0;
  char* base = (char*)d_ws;
  auto carve = [&](size_t bytes) -> char* {
    char* p = base + off;
    off += (bytes + 255) & ~(size_t)255;
    return p;
  };
  __bf16* w1aT = (__bf16*)carve(128 * 128 * 2);
  __bf16* w1bT = (__bf16*)carve(128 * 128 * 2);
  __bf16* w2aT = (__bf16*)carve(128 * 128 * 2);
  __bf16* w2bT = (__bf16*)carve(128 * 128 * 2);
  __bf16* w3aT = (__bf16*)carve(128 * 128 * 2);
  __bf16* w3bT = (__bf16*)carve(128 * 128 * 2);
  __bf16* woT  = (__bf16*)carve(384 * 128 * 2);
  float* h1      = (float*)carve((size_t)N1 * 128 * 4);
  float* h2      = (float*)carve((size_t)N2 * 128 * 4);  // later reused as h2_dn
  float* h2t     = (float*)carve((size_t)N2 * 128 * 4);
  float* h3      = (float*)carve((size_t)N3 * 128 * 4);  // later reused as h3_dn
  float* h3t     = (float*)carve((size_t)N3 * 128 * 4);
  float* h4      = (float*)carve((size_t)N4 * 128 * 4);
  float* h2_on_n1 = (float*)carve((size_t)N1 * 128 * 4);
  float* h_down   = (float*)carve((size_t)N1 * 128 * 4);
  float* h3_dn = h3;   // h3 dead after h3t
  float* h2_dn = h2;   // h2 dead after h2t
  float* outp  = (float*)d_out;

  // ---- weight prep (bf16 transposed) ----
  {
    int n = 128 * 128, bs = 256, g = (n + bs - 1) / bs;
    wT_bf16_kernel<<<g, bs, 0, stream>>>(W1a, w1aT, 128, 128);
    wT_bf16_kernel<<<g, bs, 0, stream>>>(W1b, w1bT, 128, 128);
    wT_bf16_kernel<<<g, bs, 0, stream>>>(W2a, w2aT, 128, 128);
    wT_bf16_kernel<<<g, bs, 0, stream>>>(W2b, w2bT, 128, 128);
    wT_bf16_kernel<<<g, bs, 0, stream>>>(W3a, w3aT, 128, 128);
    wT_bf16_kernel<<<g, bs, 0, stream>>>(W3b, w3bT, 128, 128);
    int no = 384 * 128, go = (no + bs - 1) / bs;
    wT_bf16_kernel<<<go, bs, 0, stream>>>(Wo, woT, 384, 128);
  }

  auto mlp_grid = [](int n) { return ((n / 16) + 3) / 4; };
  auto ew_grid = [](long long threads) { return (int)((threads + 255) / 256); };

  // ---- upward pass ----
  mlp_kernel<<<mlp_grid(N1), 128, 0, stream>>>(feat, w1aT, b1a, w1bT, b1b, h1, N1);
  seg_prod2_kernel<<<ew_grid((long long)N2 * 32), 256, 0, stream>>>(h1, src12, h2, N2);
  mlp_kernel<<<mlp_grid(N2), 128, 0, stream>>>(h2, w2aT, b2a, w2bT, b2b, h2t, N2);

  // h2_on_n1 = scatter_add(h2t via src12) onto N1
  zero_kernel<<<ew_grid((long long)N1 * 128), 256, 0, stream>>>(h2_on_n1, (size_t)N1 * 128);
  scatter_add_kernel<<<ew_grid((long long)2 * N2 * 32), 256, 0, stream>>>(h2t, src12, h2_on_n1, 2 * N2);

  seg_prod2_kernel<<<ew_grid((long long)N3 * 32), 256, 0, stream>>>(h2t, src23, h3, N3);
  mlp_kernel<<<mlp_grid(N3), 128, 0, stream>>>(h3, w3aT, b3a, w3bT, b3b, h3t, N3);
  seg_prod2_kernel<<<ew_grid((long long)N4 * 32), 256, 0, stream>>>(h3t, src34, h4, N4);

  // ---- downward pass ----
  zero_kernel<<<ew_grid((long long)N3 * 128), 256, 0, stream>>>(h3_dn, (size_t)N3 * 128);
  scatter_add_kernel<<<ew_grid((long long)2 * N4 * 32), 256, 0, stream>>>(h4, src34, h3_dn, 2 * N4);

  zero_kernel<<<ew_grid((long long)N2 * 128), 256, 0, stream>>>(h2_dn, (size_t)N2 * 128);
  scatter_add_kernel<<<ew_grid((long long)2 * N3 * 32), 256, 0, stream>>>(h3_dn, src23, h2_dn, 2 * N3);

  zero_kernel<<<ew_grid((long long)N1 * 128), 256, 0, stream>>>(h_down, (size_t)N1 * 128);
  scatter_add_kernel<<<ew_grid((long long)2 * N2 * 32), 256, 0, stream>>>(h2_dn, src12, h_down, 2 * N2);

  // ---- final projection ----
  outproj_kernel<<<mlp_grid(N1), 128, 0, stream>>>(h1, h2_on_n1, h_down, woT, bo, outp, N1);
}
